// MCFN_44581760532716
// MI455X (gfx1250) — matrix operand order repaired and checked by reading, not compiled, not run on hardware
//
#include <hip/hip_runtime.h>

// ---------------- types ----------------
typedef __attribute__((ext_vector_type(16))) __bf16 v16bf;
typedef __attribute__((ext_vector_type(8)))  __bf16 v8bf;
typedef __attribute__((ext_vector_type(8)))  float  v8f;
typedef __attribute__((ext_vector_type(4)))  float  v4f;

// ---------------- model constants ----------------
#define N_INST 15000
#define HDIM   256
#define KK     10500          // ceil(0.7*15000)
#define KPADR  10512          // kk padded to 16
#define TPATH  21001          // 1 + 2k
#define TPAD   21008
#define NPOW   16384
#define CTILES (KPADR/16)     // 657

// ================= WMMA GEMM: C[M,N] = act(A[M,K] @ Bpack + bias) =============
// A: f32 row-major (converted to bf16 inline). Bpack: pre-packed bf16 operand.
// act: 0=none 1=relu 2=tanh 3=sigmoid
__global__ void __launch_bounds__(256)
gemm_bf16(const float* __restrict__ A, int M, int K,
          const __bf16* __restrict__ Bp, const float* __restrict__ bias,
          float* __restrict__ C, int Ncols, int act)
{
  const int KB = K >> 5;
  const int tilesM = (M + 15) >> 4;
  const int tilesN = Ncols >> 4;
  int wid  = threadIdx.x >> 5;
  int lane = threadIdx.x & 31;
  int tile = blockIdx.x * 8 + wid;
  if (tile >= tilesM * tilesN) return;
  int tm = tile % tilesM, tn = tile / tilesM;
  int row = tm * 16 + (lane & 15);
  if (row >= M) row = M - 1;
  const float*  arow = A + (size_t)row * K + ((lane & 16) ? 8 : 0);
  const __bf16* bp   = Bp + (size_t)tn * KB * 512 + lane * 16;
  v8f acc = {};
  for (int kb = 0; kb < KB; ++kb) {
    const float* ap = arow + kb * 32;
    __builtin_prefetch((const void*)(ap + 256), 0, 1);
    v4f a0 = *(const v4f*)(ap);
    v4f a1 = *(const v4f*)(ap + 4);
    v4f a2 = *(const v4f*)(ap + 16);
    v4f a3 = *(const v4f*)(ap + 20);
    v16bf av;
    av[0]=(__bf16)a0[0]; av[1]=(__bf16)a0[1]; av[2]=(__bf16)a0[2]; av[3]=(__bf16)a0[3];
    av[4]=(__bf16)a1[0]; av[5]=(__bf16)a1[1]; av[6]=(__bf16)a1[2]; av[7]=(__bf16)a1[3];
    av[8]=(__bf16)a2[0]; av[9]=(__bf16)a2[1]; av[10]=(__bf16)a2[2]; av[11]=(__bf16)a2[3];
    av[12]=(__bf16)a3[0]; av[13]=(__bf16)a3[1]; av[14]=(__bf16)a3[2]; av[15]=(__bf16)a3[3];
    v16bf bv = *(const v16bf*)(bp + kb * 512);
    acc = __builtin_amdgcn_wmma_f32_16x16x32_bf16(false, av, false, bv,
                                                  (short)0, acc, false, false);
  }
  int n = tn * 16 + (lane & 15);
  float bsv = bias ? bias[n] : 0.f;
  int mbase = tm * 16 + ((lane & 16) ? 8 : 0);
  for (int r = 0; r < 8; ++r) {
    int m = mbase + r;
    if (m < M) {
      float v = acc[r] + bsv;
      if      (act == 1) v = fmaxf(v, 0.f);
      else if (act == 2) v = tanhf(v);
      else if (act == 3) v = 1.f / (1.f + __expf(-v));
      C[(size_t)m * Ncols + n] = v;
    }
  }
}

// Pack W f32 [K,Ncols] into WMMA B-operand order (lane-major 32B chunks)
__global__ void __launch_bounds__(256)
pack_b(const float* __restrict__ W, __bf16* __restrict__ P, int K, int Ncols)
{
  int total = K * Ncols;
  int KB = K >> 5;
  for (int t = blockIdx.x * blockDim.x + threadIdx.x; t < total;
       t += gridDim.x * blockDim.x) {
    int e    = t & 15;
    int lane = (t >> 4) & 31;
    int blk  = t >> 9;            // tn*KB + kb
    int kb = blk % KB, tn = blk / KB;
    int k = kb * 32 + ((lane & 16) ? 8 : 0) + (e & 7) + ((e & 8) ? 16 : 0);
    int n = tn * 16 + (lane & 15);
    P[t] = (__bf16)W[(size_t)k * Ncols + n];
  }
}

// ================= small building blocks =================
__global__ void __launch_bounds__(256)
omic_embed(const float* __restrict__ x, const float* __restrict__ W,
           const float* __restrict__ b, float* __restrict__ out, int sz)
{
  int c = threadIdx.x;
  float s = b[c];
  for (int i = 0; i < sz; ++i) s += x[i] * W[(size_t)i * HDIM + c];
  out[c] = s > 0.f ? s : __expf(s) - 1.f;   // elu
}

// Y[r] = act(X[r] @ W + b), X rows of 256, W [256,256]
__global__ void __launch_bounds__(256)
small_mm(const float* __restrict__ X, const float* __restrict__ W,
         const float* __restrict__ b, float* __restrict__ Y, int act)
{
  int r = blockIdx.x, c = threadIdx.x;
  const float* x = X + (size_t)r * HDIM;
  float s = b ? b[c] : 0.f;
  for (int k = 0; k < HDIM; ++k) s += x[k] * W[(size_t)k * HDIM + c];
  if (act == 1) s = fmaxf(s, 0.f);
  Y[(size_t)r * HDIM + c] = s;
}

__global__ void __launch_bounds__(256)
ln_kernel(const float* __restrict__ X, const float* __restrict__ g,
          const float* __restrict__ b, float* __restrict__ Y, int M)
{
  int r = blockIdx.x;
  if (r >= M) return;
  int c = threadIdx.x;
  float x = X[(size_t)r * HDIM + c];
  __shared__ float sm[256];
  sm[c] = x; __syncthreads();
  for (int s = 128; s > 0; s >>= 1) { if (c < s) sm[c] += sm[c + s]; __syncthreads(); }
  float mean = sm[0] * (1.f / 256.f); __syncthreads();
  float d = x - mean;
  sm[c] = d * d; __syncthreads();
  for (int s = 128; s > 0; s >>= 1) { if (c < s) sm[c] += sm[c + s]; __syncthreads(); }
  float var = sm[0] * (1.f / 256.f);
  Y[(size_t)r * HDIM + c] = d * rsqrtf(var + 1e-5f) * g[c] + b[c];
}

// ================= co-attention =================
// per-instance softmax over 6 omic tokens, 8 heads of d=32; also cls_attn
__global__ void __launch_bounds__(256)
attn_kernel(const float* __restrict__ Q, const float* __restrict__ Kp,
            float* __restrict__ ATT, float* __restrict__ CLS, int N)
{
  int n = blockIdx.x * blockDim.x + threadIdx.x;
  if (n >= N) return;
  const float* q = Q + (size_t)n * HDIM;
  const float scale = 0.17677669529663687f;   // 1/sqrt(32)
  float clssum = 0.f;
  for (int h = 0; h < 8; ++h) {
    float lg[6]; float mx = -3.4e38f;
    const float* qh = q + h * 32;
    for (int m = 0; m < 6; ++m) {
      const float* kr = Kp + m * HDIM + h * 32;
      float d = 0.f;
      for (int c = 0; c < 32; ++c) d += qh[c] * kr[c];
      lg[m] = d * scale;
      mx = fmaxf(mx, lg[m]);
    }
    float Z = 0.f;
    for (int m = 0; m < 6; ++m) { lg[m] = __expf(lg[m] - mx); Z += lg[m]; }
    float invZ = 1.f / Z;
    for (int m = 0; m < 6; ++m) {
      float a = lg[m] * invZ;
      ATT[(size_t)n * 48 + h * 6 + m] = a;
      clssum += a;
    }
  }
  CLS[n] = clssum * (1.f / 48.f);
}

// O2[m,hc] = V2[m,hc] + sum_n A[h][n][m] * V1[n,hc]   (48 blocks)
__global__ void __launch_bounds__(256)
o2_kernel(const float* __restrict__ ATT, const float* __restrict__ V1,
          const float* __restrict__ V2, float* __restrict__ O2, int N)
{
  int mh = blockIdx.x;
  int m = mh / 8, h = mh % 8;
  int c = threadIdx.x & 31;
  int t = threadIdx.x >> 5;
  float acc = 0.f;
  for (int n = t; n < N; n += 8)
    acc += ATT[(size_t)n * 48 + h * 6 + m] * V1[(size_t)n * HDIM + h * 32 + c];
  __shared__ float sm[256];
  sm[threadIdx.x] = acc; __syncthreads();
  if (t == 0) {
    float s = 0.f;
    for (int i = 0; i < 8; ++i) s += sm[i * 32 + c];
    O2[m * HDIM + h * 32 + c] = V2[m * HDIM + h * 32 + c] + s;
  }
}

// V1 <- V1 + A @ V2   (in place, becomes O1)
__global__ void __launch_bounds__(256)
o1_kernel(float* __restrict__ V1, const float* __restrict__ ATT,
          const float* __restrict__ V2, int N)
{
  int n = blockIdx.x;
  if (n >= N) return;
  int c = threadIdx.x;
  int h = c >> 5;
  const float* a = ATT + (size_t)n * 48 + h * 6;
  float s = 0.f;
  for (int m = 0; m < 6; ++m) s += a[m] * V2[m * HDIM + c];
  V1[(size_t)n * HDIM + c] += s;
}

// ================= top-k via bitonic sort (desc key, asc idx) =================
__global__ void __launch_bounds__(256)
sort_init(float* __restrict__ keys, int* __restrict__ idx, int N, int NP)
{
  int i = blockIdx.x * blockDim.x + threadIdx.x;
  if (i >= NP) return;
  if (i >= N) keys[i] = -3.4e38f;
  idx[i] = i;
}

__global__ void __launch_bounds__(256)
bitonic_step(float* __restrict__ key, int* __restrict__ idx,
             unsigned j, unsigned k2)
{
  unsigned i = blockIdx.x * 256u + threadIdx.x;
  unsigned ixj = i ^ j;
  if (ixj <= i) return;
  float ka = key[i], kb = key[ixj];
  int   ia = idx[i], ib = idx[ixj];
  bool up = ((i & k2) == 0u);
  bool jFirst = (kb > ka) || (kb == ka && ib < ia);
  bool iFirst = (ka > kb) || (ka == kb && ia < ib);
  if (up ? jFirst : iFirst) { key[i]=kb; key[ixj]=ka; idx[i]=ib; idx[ixj]=ia; }
}

// x_inatten partials: sum over ranks i in [kk,N) of key[i] * O1F[perm[i]]
__global__ void __launch_bounds__(256)
xin_partial(const float* __restrict__ keys, const int* __restrict__ perm,
            const float* __restrict__ O1F, float* __restrict__ part,
            int kk, int N)
{
  int c = threadIdx.x;
  float acc = 0.f;
  for (int i = kk + blockIdx.x; i < N; i += gridDim.x)
    acc += keys[i] * O1F[(size_t)perm[i] * HDIM + c];
  part[blockIdx.x * 256 + c] = acc;
}

__global__ void __launch_bounds__(256)
part_reduce(const float* __restrict__ part, float* __restrict__ out, int nb)
{
  int c = threadIdx.x;
  float s = 0.f;
  for (int b = 0; b < nb; ++b) s += part[b * 256 + c];
  out[c] = s;
}

// normalized top-k vectors in bf16 (row-major [KPADR,256]; pad rows zero)
__global__ void __launch_bounds__(256)
gather_nv(const int* __restrict__ perm, const float* __restrict__ O1F,
          __bf16* __restrict__ NVB, int kk)
{
  int i = blockIdx.x;
  int c = threadIdx.x;
  float v = (i < kk) ? O1F[(size_t)perm[i] * HDIM + c] : 0.f;
  __shared__ float sm[256];
  sm[c] = v * v; __syncthreads();
  for (int s = 128; s > 0; s >>= 1) { if (c < s) sm[c] += sm[c + s]; __syncthreads(); }
  float nrm = sqrtf(sm[0]);
  float inv = (i < kk && nrm > 0.f) ? 1.f / nrm : 0.f;
  NVB[(size_t)i * HDIM + c] = (__bf16)(v * inv);
}

// argmax_j!=i of <nv_i, nv_j> using WMMA, no materialized cos matrix
__global__ void __launch_bounds__(256)
cos_argmax(const __bf16* __restrict__ NVB, int* __restrict__ MAXL,
           int kk, int tiles)
{
  int rt = blockIdx.x;
  int wid = threadIdx.x >> 5, lane = threadIdx.x & 31;
  int arow = rt * 16 + (lane & 15);
  const __bf16* ap = NVB + (size_t)arow * HDIM + ((lane & 16) ? 16 : 0);
  v16bf A[8];
  for (int s = 0; s < 8; ++s) {
    v8bf c1 = *(const v8bf*)(ap + s * 32);
    v8bf c2 = *(const v8bf*)(ap + s * 32 + 8);
    v16bf av;
    for (int e = 0; e < 8; ++e) { av[e] = c1[e]; av[8 + e] = c2[e]; }
    A[s] = av;
  }
  float best[8]; int bidx[8];
  for (int r = 0; r < 8; ++r) { best[r] = -3.4e38f; bidx[r] = 0; }
  int grow_base = rt * 16 + ((lane & 16) ? 8 : 0);
  int ncol = lane & 15;
  for (int ct = wid; ct < tiles; ct += 8) {
    const __bf16* bp = NVB + (size_t)(ct * 16 + ncol) * HDIM + ((lane & 16) ? 16 : 0);
    v8f acc = {};
    for (int s = 0; s < 8; ++s) {
      v16bf bv = *(const v16bf*)(bp + s * 32);
      acc = __builtin_amdgcn_wmma_f32_16x16x32_bf16(false, A[s], false, bv,
                                                    (short)0, acc, false, false);
    }
    int gn = ct * 16 + ncol;
    for (int r = 0; r < 8; ++r) {
      int gm = grow_base + r;
      float v = acc[r];
      if (gn == gm || gn >= kk) v = -3.4e38f;
      if (v > best[r]) { best[r] = v; bidx[r] = gn; }   // gn increasing -> first-max
    }
  }
  __shared__ float smax[16][257];
  __shared__ int   sarg[16][257];
  for (int r = 0; r < 16; ++r) { smax[r][threadIdx.x] = -3.4e38f; sarg[r][threadIdx.x] = 0; }
  __syncthreads();
  int rrbase = (lane & 16) ? 8 : 0;
  for (int r = 0; r < 8; ++r) {
    smax[rrbase + r][threadIdx.x] = best[r];
    sarg[rrbase + r][threadIdx.x] = bidx[r];
  }
  __syncthreads();
  if (threadIdx.x < 16) {
    int row = threadIdx.x;
    float bv = -3.4e38f; int bi = 0;
    for (int t = 0; t < 256; ++t) {
      float v = smax[row][t]; int ix = sarg[row][t];
      if (v > bv || (v == bv && ix < bi)) { bv = v; bi = ix; }
    }
    int gr = rt * 16 + row;
    if (gr < kk) MAXL[gr] = bi;
  }
}

// rows 1..k = vectors ; rows k+1..2k = 0.5*(v_i + v_maxlist[i])
__global__ void __launch_bounds__(256)
assemble_xpath(const int* __restrict__ perm, const float* __restrict__ O1F,
               const int* __restrict__ MAXL, float* __restrict__ XP, int kk)
{
  int i = blockIdx.x;
  int c = threadIdx.x;
  float v = O1F[(size_t)perm[i] * HDIM + c];
  XP[(size_t)(1 + i) * HDIM + c] = v;
  int j = MAXL[i];
  float w = O1F[(size_t)perm[j] * HDIM + c];
  XP[(size_t)(1 + kk + i) * HDIM + c] = 0.5f * (v + w);
}

// gated attention score: sc = (tanh*sig) @ W_pc + b_pc
__global__ void __launch_bounds__(256)
scores_kernel(const float* __restrict__ Aa, const float* __restrict__ Ss,
              const float* __restrict__ Wc, const float* __restrict__ bc,
              float* __restrict__ SC, int M)
{
  int n = blockIdx.x * blockDim.x + threadIdx.x;
  if (n >= M) return;
  float s = bc[0];
  for (int j = 0; j < 64; ++j) s += Aa[(size_t)n * 64 + j] * Ss[(size_t)n * 64 + j] * Wc[j];
  SC[n] = s;
}

__global__ void __launch_bounds__(256)
softmax_stats(const float* __restrict__ SC, float* __restrict__ red, int M)
{
  int t = threadIdx.x;
  __shared__ float sm[256];
  float mx = -3.4e38f;
  for (int i = t; i < M; i += 256) mx = fmaxf(mx, SC[i]);
  sm[t] = mx; __syncthreads();
  for (int s = 128; s > 0; s >>= 1) { if (t < s) sm[t] = fmaxf(sm[t], sm[t + s]); __syncthreads(); }
  float m = sm[0]; __syncthreads();
  float z = 0.f;
  for (int i = t; i < M; i += 256) z += __expf(SC[i] - m);
  sm[t] = z; __syncthreads();
  for (int s = 128; s > 0; s >>= 1) { if (t < s) sm[t] += sm[t + s]; __syncthreads(); }
  if (t == 0) { red[0] = m; red[1] = sm[0]; }
}

__global__ void __launch_bounds__(256)
pool_partial(const float* __restrict__ SC, const float* __restrict__ red,
             const float* __restrict__ X, float* __restrict__ part, int M)
{
  int c = threadIdx.x;
  float m = red[0], invz = 1.f / red[1];
  float acc = 0.f;
  for (int n = blockIdx.x; n < M; n += gridDim.x)
    acc += __expf(SC[n] - m) * invz * X[(size_t)n * HDIM + c];
  part[blockIdx.x * 256 + c] = acc;
}

// ================= tail (all tiny stages, single block) =================
struct LnRes { float d, inv; };
__device__ inline LnRes ln_stats(float x, float* red, int t)
{
  red[t] = x; __syncthreads();
  for (int s = 128; s > 0; s >>= 1) { if (t < s) red[t] += red[t + s]; __syncthreads(); }
  float mean = red[0] * (1.f / 256.f); __syncthreads();
  float d = x - mean;
  red[t] = d * d; __syncthreads();
  for (int s = 128; s > 0; s >>= 1) { if (t < s) red[t] += red[t + s]; __syncthreads(); }
  float var = red[0] * (1.f / 256.f); __syncthreads();
  LnRes r; r.d = d; r.inv = rsqrtf(var + 1e-5f);
  return r;
}

__global__ void __launch_bounds__(256)
tail_kernel(const float* POOL, const float* O2F,
            const float* W_pr2, const float* b_pr2,
            const float* g_oln, const float* b_oln,
            const float* W_oa, const float* b_oa,
            const float* W_ob, const float* b_ob,
            const float* W_oc, const float* b_oc,
            const float* W_or, const float* b_or,
            const float* W_mm1, const float* b_mm1,
            const float* W_mm2, const float* b_mm2,
            const float* g_lt, const float* b_lt,
            const float* W_tm1, const float* b_tm1,
            const float* W_tm2, const float* b_tm2,
            const float* g_lc, const float* b_lc,
            const float* W_cm1, const float* b_cm1,
            const float* W_cm2, const float* b_cm2,
            const float* g_lnf, const float* b_lnf,
            const float* W_cls, const float* b_cls,
            float* out)
{
  int t = threadIdx.x;
  __shared__ float red[256];
  __shared__ float hpg[256], hog[256], hmm[256], pool_s[256];
  __shared__ float ho[6][256];
  __shared__ float Ao[6], wsm[8];
  __shared__ float xrow[3][256], xt[3][256], cl[3][256];
  __shared__ float ecm[3][64];

  pool_s[t] = POOL[t]; __syncthreads();
  { // path projection
    float s = b_pr2[t];
    for (int k = 0; k < 256; ++k) s += pool_s[k] * W_pr2[k * 256 + t];
    hpg[t] = fmaxf(s, 0.f);
  }
  __syncthreads();
  // omic LN rows
  for (int m = 0; m < 6; ++m) {
    LnRes r = ln_stats(O2F[m * 256 + t], red, t);
    ho[m][t] = r.d * r.inv * g_oln[t] + b_oln[t];
    __syncthreads();
  }
  // omic gated attention logits
  for (int m = 0; m < 6; ++m) {
    float p = 0.f;
    if (t < 64) {
      float a = b_oa[t], s2 = b_ob[t];
      for (int k = 0; k < 256; ++k) { float h = ho[m][k]; a += h * W_oa[k * 64 + t]; s2 += h * W_ob[k * 64 + t]; }
      p = tanhf(a) * (1.f / (1.f + __expf(-s2))) * W_oc[t];
    }
    red[t] = p; __syncthreads();
    for (int s = 128; s > 0; s >>= 1) { if (t < s) red[t] += red[t + s]; __syncthreads(); }
    if (t == 0) Ao[m] = red[0] + b_oc[0];
    __syncthreads();
  }
  if (t == 0) {
    float mx = -3.4e38f;
    for (int m = 0; m < 6; ++m) mx = fmaxf(mx, Ao[m]);
    float z = 0.f;
    for (int m = 0; m < 6; ++m) { wsm[m] = __expf(Ao[m] - mx); z += wsm[m]; }
    for (int m = 0; m < 6; ++m) wsm[m] /= z;
  }
  __syncthreads();
  { // omic pooled + projection
    float hp = 0.f;
    for (int m = 0; m < 6; ++m) hp += wsm[m] * ho[m][t];
    red[t] = hp;
  }
  __syncthreads();
  pool_s[t] = red[t]; __syncthreads();
  {
    float s = b_or[t];
    for (int k = 0; k < 256; ++k) s += pool_s[k] * W_or[k * 256 + t];
    hog[t] = fmaxf(s, 0.f);
  }
  __syncthreads();
  { // fusion MLP
    float s = b_mm1[t];
    for (int k = 0; k < 256; ++k)
      s += hpg[k] * W_mm1[k * 256 + t] + hog[k] * W_mm1[(256 + k) * 256 + t];
    red[t] = fmaxf(s, 0.f);
  }
  __syncthreads();
  {
    float s = b_mm2[t];
    for (int k = 0; k < 256; ++k) s += red[k] * W_mm2[k * 256 + t];
    hmm[t] = fmaxf(s, 0.f);
  }
  __syncthreads();
  xrow[0][t] = hpg[t]; xrow[1][t] = hog[t]; xrow[2][t] = hmm[t];
  __syncthreads();
  // token-mix: LN then [256,3]@[3,16] elu @[16,3]
  for (int r = 0; r < 3; ++r) {
    LnRes s = ln_stats(xrow[r][t], red, t);
    xt[r][t] = s.d * s.inv * g_lt[t] + b_lt[t];
    __syncthreads();
  }
  {
    float h16[16];
    for (int j = 0; j < 16; ++j) {
      float s = b_tm1[j];
      for (int r = 0; r < 3; ++r) s += xt[r][t] * W_tm1[r * 16 + j];
      h16[j] = s > 0.f ? s : __expf(s) - 1.f;
    }
    for (int r = 0; r < 3; ++r) {
      float s = b_tm2[r];
      for (int j = 0; j < 16; ++j) s += h16[j] * W_tm2[j * 3 + r];
      xrow[r][t] += s;
    }
  }
  __syncthreads();
  // channel-mix: LN, [3,256]@[256,64] elu @[64,256]
  for (int r = 0; r < 3; ++r) {
    LnRes s = ln_stats(xrow[r][t], red, t);
    cl[r][t] = s.d * s.inv * g_lc[t] + b_lc[t];
    __syncthreads();
  }
  if (t < 192) {
    int r = t / 64, j = t % 64;
    float s = b_cm1[j];
    for (int c = 0; c < 256; ++c) s += cl[r][c] * W_cm1[c * 64 + j];
    ecm[r][j] = s > 0.f ? s : __expf(s) - 1.f;
  }
  __syncthreads();
  for (int r = 0; r < 3; ++r) {
    float s = b_cm2[t];
    for (int j = 0; j < 64; ++j) s += ecm[r][j] * W_cm2[j * 256 + t];
    xrow[r][t] += s;
  }
  __syncthreads();
  // final LN, mean over 3 tokens, classifier
  for (int r = 0; r < 3; ++r) {
    LnRes s = ln_stats(xrow[r][t], red, t);
    cl[r][t] = s.d * s.inv * g_lnf[t] + b_lnf[t];
    __syncthreads();
  }
  red[t] = (cl[0][t] + cl[1][t] + cl[2][t]) * (1.f / 3.f);
  __syncthreads();
  if (t < 4) {
    float s = b_cls[t];
    for (int c = 0; c < 256; ++c) s += red[c] * W_cls[c * 4 + t];
    out[t] = s;
  }
}

// ================= host side =================
enum {
  IN_XIMG = 0,
  IN_WIMG = 7, IN_BIMG = 8,
  IN_WSIG0 = 9, IN_BSIG0 = 15,
  IN_WFCQ = 21, IN_BFCQ = 22, IN_WFCK = 23, IN_BFCK = 24,
  IN_WV1 = 25, IN_BV1 = 26, IN_WV2 = 27, IN_BV2 = 28,
  IN_GLNQ = 29, IN_BLNQ = 30, IN_GLNK = 31, IN_BLNK = 32,
  IN_WPR1 = 33, IN_BPR1 = 34, IN_GPLN = 35, IN_BPLN = 36,
  IN_WPA = 37, IN_BPA = 38, IN_WPB = 39, IN_BPB = 40, IN_WPC = 41, IN_BPC = 42,
  IN_WPR2 = 43, IN_BPR2 = 44,
  IN_GOLN = 45, IN_BOLN = 46, IN_WOA = 47, IN_BOA = 48,
  IN_WOB = 49, IN_BOB = 50, IN_WOC = 51, IN_BOC = 52,
  IN_WOR = 53, IN_BOR = 54,
  IN_WMM1 = 55, IN_BMM1 = 56, IN_WMM2 = 57, IN_BMM2 = 58,
  IN_GLT = 59, IN_BLT = 60, IN_WTM1 = 61, IN_BTM1 = 62, IN_WTM2 = 63, IN_BTM2 = 64,
  IN_GLC = 65, IN_BLC = 66, IN_WCM1 = 67, IN_BCM1 = 68, IN_WCM2 = 69, IN_BCM2 = 70,
  IN_GLNF = 71, IN_BLNF = 72, IN_WCLS = 73, IN_BCLS = 74
};

static inline size_t algn(size_t x) { return (x + 255) & ~(size_t)255; }

extern "C" void kernel_launch(void* const* d_in, const int* in_sizes, int n_in,
                              void* d_out, int out_size, void* d_ws, size_t ws_size,
                              hipStream_t stream)
{
  (void)n_in; (void)out_size; (void)ws_size;
  char* ws = (char*)d_ws;

  const size_t SZ_BIG   = (size_t)15008 * HDIM * 4;
  const size_t OFF_HP   = 0;
  const size_t OFF_Q    = OFF_HP + SZ_BIG;
  const size_t OFF_V1   = OFF_Q + SZ_BIG;
  const size_t OFF_ATT  = OFF_V1 + SZ_BIG;
  const size_t OFF_CLS  = OFF_ATT + (size_t)15008 * 48 * 4;
  const size_t OFF_PERM = OFF_CLS + (size_t)NPOW * 4;
  const size_t OFF_NVB  = OFF_PERM + (size_t)NPOW * 4;
  const size_t OFF_MAXL = OFF_NVB + (size_t)KPADR * HDIM * 2;
  const size_t OFF_XP   = algn(OFF_MAXL + (size_t)KPADR * 4);
  const size_t OFF_HPP  = OFF_XP + (size_t)TPAD * HDIM * 4;
  const size_t OFF_SC   = OFF_HPP + (size_t)TPAD * HDIM * 4;
  const size_t OFF_SMALL= algn(OFF_SC + (size_t)TPAD * 4);
  const size_t OFF_PART = OFF_SMALL + 65536;
  const size_t OFF_PACK = OFF_PART + (size_t)128 * 256 * 4;

  float* HP   = (float*)(ws + OFF_HP);
  float* Qb   = (float*)(ws + OFF_Q);
  float* V1b  = (float*)(ws + OFF_V1);
  float* ATTb = (float*)(ws + OFF_ATT);
  float* CLSb = (float*)(ws + OFF_CLS);
  int*   PERMb= (int*)  (ws + OFF_PERM);
  __bf16* NVBb= (__bf16*)(ws + OFF_NVB);
  int*   MAXLb= (int*)  (ws + OFF_MAXL);
  float* XPb  = (float*)(ws + OFF_XP);
  float* HPPb = (float*)(ws + OFF_HPP);
  float* SCb  = (float*)(ws + OFF_SC);
  float* SMALL= (float*)(ws + OFF_SMALL);
  float* PART = (float*)(ws + OFF_PART);

  // aliases (temporally dead buffers reused)
  float* LNO  = Qb;                              // ln(O1)
  float* O1F  = HP;                              // final O1 projection
  float* HPLN = XPb;                             // ln(hp) overwrites x_path
  float* AS_A = V1b;                             // tanh branch [TPAD,64]
  float* AS_S = V1b + (size_t)TPAD * 64;         // sigmoid branch

  float* HOM  = SMALL + 0;      // [6,256]
  float* KPs  = SMALL + 1536;
  float* V2s  = SMALL + 3072;
  float* O2s  = SMALL + 4608;
  float* O2LN = SMALL + 6144;
  float* O2Fs = SMALL + 7680;
  float* POOLs= SMALL + 9216;   // [256]
  float* REDs = SMALL + 9472;   // [2]

  __bf16* PK_IMG = (__bf16*)(ws + OFF_PACK);
  __bf16* PK_FCQ = PK_IMG + (size_t)1024 * 256;
  __bf16* PK_V1  = PK_FCQ + (size_t)256 * 256;
  __bf16* PK_PR1 = PK_V1  + (size_t)256 * 256;
  __bf16* PK_PA  = PK_PR1 + (size_t)256 * 256;
  __bf16* PK_PB  = PK_PA  + (size_t)256 * 64;

  auto F = [&](int i) { return (const float*)d_in[i]; };
  auto gB = [](int M, int Ncols) {
    int t = ((M + 15) / 16) * (Ncols / 16);
    return (t + 7) / 8;
  };

  // ---- pack weights into WMMA B-operand layout ----
  pack_b<<<256, 256, 0, stream>>>(F(IN_WIMG), PK_IMG, 1024, 256);
  pack_b<<<64, 256, 0, stream>>>(F(IN_WFCQ), PK_FCQ, 256, 256);
  pack_b<<<64, 256, 0, stream>>>(F(IN_WV1),  PK_V1,  256, 256);
  pack_b<<<64, 256, 0, stream>>>(F(IN_WPR1), PK_PR1, 256, 256);
  pack_b<<<16, 256, 0, stream>>>(F(IN_WPA),  PK_PA,  256, 64);
  pack_b<<<16, 256, 0, stream>>>(F(IN_WPB),  PK_PB,  256, 64);

  // ---- embeddings ----
  for (int j = 0; j < 6; ++j)
    omic_embed<<<1, 256, 0, stream>>>(F(1 + j), F(IN_WSIG0 + j), F(IN_BSIG0 + j),
                                      HOM + j * 256, in_sizes[1 + j]);
  gemm_bf16<<<gB(N_INST, 256), 256, 0, stream>>>(F(IN_XIMG), N_INST, 1024,
                                                 PK_IMG, F(IN_BIMG), HP, 256, 1);

  // ---- co-attention block ----
  small_mm<<<6, 256, 0, stream>>>(HOM, F(IN_WFCK), F(IN_BFCK), KPs, 1);   // Kp
  small_mm<<<6, 256, 0, stream>>>(KPs, F(IN_WV2), F(IN_BV2), V2s, 0);     // V2
  gemm_bf16<<<gB(N_INST, 256), 256, 0, stream>>>(HP, N_INST, 256, PK_FCQ,
                                                 F(IN_BFCQ), Qb, 256, 1); // Q
  gemm_bf16<<<gB(N_INST, 256), 256, 0, stream>>>(Qb, N_INST, 256, PK_V1,
                                                 F(IN_BV1), V1b, 256, 0); // V1
  attn_kernel<<<(N_INST + 255) / 256, 256, 0, stream>>>(Qb, KPs, ATTb, CLSb, N_INST);
  o2_kernel<<<48, 256, 0, stream>>>(ATTb, V1b, V2s, O2s, N_INST);
  o1_kernel<<<N_INST, 256, 0, stream>>>(V1b, ATTb, V2s, N_INST);          // V1 -> O1
  ln_kernel<<<N_INST, 256, 0, stream>>>(V1b, F(IN_GLNQ), F(IN_BLNQ), LNO, N_INST);
  gemm_bf16<<<gB(N_INST, 256), 256, 0, stream>>>(LNO, N_INST, 256, PK_FCQ,
                                                 F(IN_BFCQ), O1F, 256, 1); // O1 final
  ln_kernel<<<6, 256, 0, stream>>>(O2s, F(IN_GLNK), F(IN_BLNK), O2LN, 6);
  small_mm<<<6, 256, 0, stream>>>(O2LN, F(IN_WFCK), F(IN_BFCK), O2Fs, 1);  // O2 final

  // ---- top-k (bitonic sort: key desc, idx asc) ----
  sort_init<<<NPOW / 256, 256, 0, stream>>>(CLSb, PERMb, N_INST, NPOW);
  for (unsigned k2 = 2; k2 <= (unsigned)NPOW; k2 <<= 1)
    for (unsigned j = k2 >> 1; j > 0; j >>= 1)
      bitonic_step<<<NPOW / 256, 256, 0, stream>>>(CLSb, PERMb, j, k2);

  // ---- ODA ----
  xin_partial<<<32, 256, 0, stream>>>(CLSb, PERMb, O1F, PART, KK, N_INST);
  part_reduce<<<1, 256, 0, stream>>>(PART, XPb, 32);                 // x_path row 0
  gather_nv<<<KPADR, 256, 0, stream>>>(PERMb, O1F, NVBb, KK);
  cos_argmax<<<CTILES, 256, 0, stream>>>(NVBb, MAXLb, KK, CTILES);
  assemble_xpath<<<KK, 256, 0, stream>>>(PERMb, O1F, MAXLb, XPb, KK);

  // ---- path branch ----
  gemm_bf16<<<gB(TPATH, 256), 256, 0, stream>>>(XPb, TPATH, 256, PK_PR1,
                                                F(IN_BPR1), HPPb, 256, 1);
  ln_kernel<<<TPATH, 256, 0, stream>>>(HPPb, F(IN_GPLN), F(IN_BPLN), HPLN, TPATH);
  gemm_bf16<<<gB(TPATH, 64), 256, 0, stream>>>(HPLN, TPATH, 256, PK_PA,
                                               F(IN_BPA), AS_A, 64, 2);
  gemm_bf16<<<gB(TPATH, 64), 256, 0, stream>>>(HPLN, TPATH, 256, PK_PB,
                                               F(IN_BPB), AS_S, 64, 3);
  scores_kernel<<<(TPATH + 255) / 256, 256, 0, stream>>>(AS_A, AS_S, F(IN_WPC),
                                                         F(IN_BPC), SCb, TPATH);
  softmax_stats<<<1, 256, 0, stream>>>(SCb, REDs, TPATH);
  pool_partial<<<128, 256, 0, stream>>>(SCb, REDs, HPLN, PART, TPATH);
  part_reduce<<<1, 256, 0, stream>>>(PART, POOLs, 128);

  // ---- everything tiny, fused ----
  tail_kernel<<<1, 256, 0, stream>>>(
      POOLs, O2Fs,
      F(IN_WPR2), F(IN_BPR2),
      F(IN_GOLN), F(IN_BOLN),
      F(IN_WOA), F(IN_BOA), F(IN_WOB), F(IN_BOB), F(IN_WOC), F(IN_BOC),
      F(IN_WOR), F(IN_BOR),
      F(IN_WMM1), F(IN_BMM1), F(IN_WMM2), F(IN_BMM2),
      F(IN_GLT), F(IN_BLT), F(IN_WTM1), F(IN_BTM1), F(IN_WTM2), F(IN_BTM2),
      F(IN_GLC), F(IN_BLC), F(IN_WCM1), F(IN_BCM1), F(IN_WCM2), F(IN_BCM2),
      F(IN_GLNF), F(IN_BLNF), F(IN_WCLS), F(IN_BCLS),
      (float*)d_out);
}
